// Capsule_16449724744267
// MI455X (gfx1250) — compile-verified, matrix operand order checked by reading, not run on hardware
//
#include <hip/hip_runtime.h>
#include <math.h>

// CDNA5 / gfx1250: wave32, WMMA 16x16x32 f16 -> f32, async global->LDS staging
typedef __attribute__((ext_vector_type(16))) _Float16 v16h;
typedef __attribute__((ext_vector_type(8)))  float    v8f;

#define N_IN   1024
#define DIN    128            // inner contraction dim of the conv1d GEMM
#define NCAP   32
#define DC     16
#define NK     512            // NCAP*DC
#define CHUNK  64             // i-rows staged per step
#define NCHUNK (N_IN / CHUNK) // 16
#define EPSF   1e-7f

// 16x32 f16 WMMA fragment (A-style; B uses same lane pattern with row = N col).
// 16 f16 per lane as two contiguous 16B chunks:
// elems[0..7] = K k0+half*8.. ; elems[8..15] = K k0+16+half*8..
__device__ __forceinline__ v16h frag_ld(const _Float16* __restrict__ base,
                                        int row, int stride, int k0, int hl) {
  const _Float16* p0 = base + row * stride + k0 + hl * 8;
  const _Float16* p1 = p0 + 16;
  v16h f;
#pragma unroll
  for (int j = 0; j < 8; ++j) { f[j] = p0[j]; f[j + 8] = p1[j]; }
  return f;
}

// Issue async copy of one 32KB chunk (64x128 f32) global -> LDS staging.
// 2048 16-byte units over 256 threads = 8 issues/thread, ASYNCcnt-tracked.
__device__ __forceinline__ void issue_async_chunk(const float* __restrict__ gsrc,
                                                  unsigned lds_base, int t) {
#pragma unroll
  for (int e = 0; e < 8; ++e) {
    int u = e * 256 + t;
    unsigned la = lds_base + (unsigned)u * 16u;
    unsigned long long ga = (unsigned long long)(size_t)gsrc
                          + (unsigned long long)u * 16ull;
    asm volatile("global_load_async_to_lds_b128 %0, %1, off"
                 :: "v"(la), "v"(ga) : "memory");
  }
}

__device__ __forceinline__ void wait_async0() {
  asm volatile("s_wait_asynccnt 0x0" ::: "memory");
}

__global__ __launch_bounds__(256, 2)
void capsule_routing_kernel(const float* __restrict__ inp,
                            const float* __restrict__ kern,
                            float* __restrict__ out) {
  __shared__ _Float16 sh_in   [CHUNK * DIN];   // chunk [i][k] f16      16KB
  __shared__ _Float16 sh_inT  [DIN * CHUNK];   // transposed [k][i]     16KB
  __shared__ float    sh_stage[CHUNK * DIN];   // async f32 staging     32KB
  __shared__ float    sh_b1   [CHUNK * NCAP];  // routing logits         8KB
  __shared__ _Float16 sh_cT   [NCAP * CHUNK];  // softmax coeffs [n][i]  4KB
  __shared__ _Float16 sh_w    [NCAP * DIN];    // w[n][k]=kernel@o       8KB
  __shared__ float    sh_t    [NCAP * DIN];    // t[n][k]=c@inputs      16KB
  __shared__ float    sh_o    [NCAP * DC];     // capsule outputs        2KB
  __shared__ float    sh_s    [2 * DIN];       // column-sum partials    1KB

  const int b    = blockIdx.x;
  const int t    = threadIdx.x;
  const int wv   = t >> 5;          // wave id 0..7
  const int lane = t & 31;
  const int l16  = lane & 15;
  const int hl   = lane >> 4;       // lane half
  const int mt1  = wv & 3;          // GEMM1 M-tile
  const int nt1  = wv >> 2;         // GEMM1 N-tile
  const float* in_b = inp + (size_t)b * N_IN * DIN;
  const unsigned stage_lds = (unsigned)(size_t)(void*)sh_stage;

  // ---------- iter 0 collapses: s[k] = sum_i inputs[b,i,k] ----------
  {
    int k = t & (DIN - 1);
    int part = t >> 7;
    float acc = 0.f;
    for (int i = part; i < N_IN; i += 2) acc += in_b[i * DIN + k];
    sh_s[part * DIN + k] = acc;
  }
  __syncthreads();
  if (t < DIN) sh_s[t] += sh_s[DIN + t];
  __syncthreads();
  // o0[n,d] = (1/32) * s . kernel[:, n*16+d]
#pragma unroll
  for (int rep = 0; rep < 2; ++rep) {
    int idx = t + rep * 256, n = idx >> 4, d = idx & 15;
    float acc = 0.f;
#pragma unroll 4
    for (int k = 0; k < DIN; ++k) acc += sh_s[k] * kern[k * NK + n * DC + d];
    sh_o[idx] = acc * (1.0f / 32.0f);
  }
  __syncthreads();
  if (t < NCAP) {                                    // squash
    float s2 = EPSF;
#pragma unroll
    for (int d = 0; d < DC; ++d) { float v = sh_o[t * DC + d]; s2 += v * v; }
    float sc = sqrtf(s2) / (0.5f + s2);
#pragma unroll
    for (int d = 0; d < DC; ++d) sh_o[t * DC + d] *= sc;
  }

  // ---------- routing iterations 1 and 2 ----------
  for (int it = 0; it < 2; ++it) {
    __syncthreads();
    // w[n][k] = sum_d kernel[k, n*16+d] * o[n,d]  (f16 B-matrix of GEMM1)
#pragma unroll 2
    for (int rep = 0; rep < 16; ++rep) {
      int idx = t + rep * 256, n = idx >> 7, k = idx & 127;
      float acc = 0.f;
#pragma unroll
      for (int d = 0; d < DC; ++d)
        acc += kern[k * NK + n * DC + d] * sh_o[n * DC + d];
      sh_w[n * DIN + k] = (_Float16)acc;
    }
    __syncthreads();
    issue_async_chunk(in_b, stage_lds, t);           // prefetch chunk 0

    v8f tacc0 = {}, tacc1 = {};   // persistent t[n][k] tile accumulators
    for (int c = 0; c < NCHUNK; ++c) {
      wait_async0();              // this wave's staging copies landed
      __syncthreads();            // everyone's landed; prev GEMM2 reads done
      // f32 staging -> f16 row-major + transposed
#pragma unroll 4
      for (int e = 0; e < 32; ++e) {
        int idx = e * 256 + t, i = idx >> 7, k = idx & 127;
        _Float16 v = (_Float16)sh_stage[idx];
        sh_in [i * DIN + k]   = v;
        sh_inT[k * CHUNK + i] = v;
      }
      __syncthreads();            // staging consumed, f16 tiles ready
      if (c + 1 < NCHUNK)         // overlap next chunk's copy with compute
        issue_async_chunk(in_b + (size_t)(c + 1) * CHUNK * DIN, stage_lds, t);

      // GEMM1: logits[i][n] = inputs_chunk @ w^T ; wave owns tile (mt1,nt1)
      v8f a0 = {};
#pragma unroll
      for (int ks = 0; ks < 4; ++ks) {
        v16h af = frag_ld(sh_in, 16 * mt1 + l16, DIN, 32 * ks, hl);
        v16h bf = frag_ld(sh_w,  16 * nt1 + l16, DIN, 32 * ks, hl);
        a0 = __builtin_amdgcn_wmma_f32_16x16x32_f16(false, af, false, bf,
                                                    (short)0, a0, false, false);
      }
#pragma unroll
      for (int r = 0; r < 8; ++r)
        sh_b1[(16 * mt1 + r + 8 * hl) * NCAP + 16 * nt1 + l16] = a0[r];
      __syncthreads();
      // softmax over capsules n for each row i (threads 0..63)
      if (t < CHUNK) {
        float mx = -3.4e38f;
        for (int n = 0; n < NCAP; ++n) mx = fmaxf(mx, sh_b1[t * NCAP + n]);
        float sum = 0.f;
        for (int n = 0; n < NCAP; ++n) sum += __expf(sh_b1[t * NCAP + n] - mx);
        float inv = 1.0f / sum;
        for (int n = 0; n < NCAP; ++n)
          sh_cT[n * CHUNK + t] = (_Float16)(__expf(sh_b1[t * NCAP + n] - mx) * inv);
      }
      __syncthreads();
      // GEMM2: t[n][k] += c^T @ inputs_chunk ; wave owns N-tile wv (k dim)
#pragma unroll
      for (int ks = 0; ks < 2; ++ks) {
        v16h bf = frag_ld(sh_inT, 16 * wv + l16, CHUNK, 32 * ks, hl);
        v16h c0 = frag_ld(sh_cT,  l16,           CHUNK, 32 * ks, hl);
        v16h c1 = frag_ld(sh_cT,  16 + l16,      CHUNK, 32 * ks, hl);
        tacc0 = __builtin_amdgcn_wmma_f32_16x16x32_f16(false, c0, false, bf,
                                                       (short)0, tacc0, false, false);
        tacc1 = __builtin_amdgcn_wmma_f32_16x16x32_f16(false, c1, false, bf,
                                                       (short)0, tacc1, false, false);
      }
    }
    __syncthreads();
#pragma unroll
    for (int r = 0; r < 8; ++r) {
      int m = r + 8 * hl;
      sh_t[ m       * DIN + 16 * wv + l16] = tacc0[r];
      sh_t[(16 + m) * DIN + 16 * wv + l16] = tacc1[r];
    }
    __syncthreads();
    // o[n,d] = t[n,:] . kernel[:, n*16+d], then squash
#pragma unroll
    for (int rep = 0; rep < 2; ++rep) {
      int idx = t + rep * 256, n = idx >> 4, d = idx & 15;
      float acc = 0.f;
#pragma unroll 4
      for (int k = 0; k < DIN; ++k)
        acc += sh_t[n * DIN + k] * kern[k * NK + n * DC + d];
      sh_o[idx] = acc;
    }
    __syncthreads();
    if (t < NCAP) {
      float s2 = EPSF;
#pragma unroll
      for (int d = 0; d < DC; ++d) { float v = sh_o[t * DC + d]; s2 += v * v; }
      float sc = sqrtf(s2) / (0.5f + s2);
#pragma unroll
      for (int d = 0; d < DC; ++d) sh_o[t * DC + d] *= sc;
    }
  }
  __syncthreads();
#pragma unroll
  for (int rep = 0; rep < 2; ++rep) {
    int idx = t + rep * 256;
    out[(size_t)b * NK + idx] = sh_o[idx];
  }
}

extern "C" void kernel_launch(void* const* d_in, const int* in_sizes, int n_in,
                              void* d_out, int out_size, void* d_ws, size_t ws_size,
                              hipStream_t stream) {
  (void)n_in; (void)out_size; (void)d_ws; (void)ws_size;
  const float* inp  = (const float*)d_in[0];
  const float* kern = (const float*)d_in[1];
  float* out = (float*)d_out;
  int batches = in_sizes[0] / (N_IN * DIN);   // 128
  hipLaunchKernelGGL(capsule_routing_kernel, dim3(batches), dim3(256), 0, stream,
                     inp, kern, out);
}
